// QFLayer_50302656971474
// MI455X (gfx1250) — matrix operand order repaired and checked by reading, not compiled, not run on hardware
//
#include <hip/hip_runtime.h>
#include <cmath>

// ---------------------------------------------------------------------------
// Fused QFLayer for MI455X (gfx1250, wave32, WMMA 16x16x32 f16 -> f32 acc).
// One workgroup (256 threads = 8 waves) per batch element; all intermediates
// in LDS; every GEMM on v_wmma_f32_16x16x32_f16 with B-fragment reuse
// (each weight fragment feeds 4 WMMAs -> one pass over W per block).
// ---------------------------------------------------------------------------

typedef __attribute__((ext_vector_type(16))) _Float16 v16h;
typedef __attribute__((ext_vector_type(8)))  _Float16 v8h;
typedef __attribute__((ext_vector_type(8)))  float    v8f;

#define B_    2048
#define S_    64
#define HID_  128
#define NH_   2
#define DH_   64
#define FF_   256
#define MAXP_ 64

// ---- LDS arena (byte offsets), phase-aliased --------------------------------
#define OFF_A16    0           // half[64][128]  raw q_repr (f16)        (A)
#define OFF_Q16    (16*1024)   // half[64][128]  Q                        (A,B)
#define OFF_K16    (32*1024)   // half[64][128]  K                        (A,B)
#define OFF_VT16   (48*1024)   // half[128][64]  V transposed             (A,B)
#define OFF_S32    (64*1024)   // float[64][64]  scores (per head)        (B)
#define OFF_P16    (80*1024)   // half[64][64]   softmax probs            (B)
#define OFF_XCAT   (88*1024)   // half[64][256]  [im_repr | ctx] f16      (B,C)
#define OFF_INTER  (120*1024)  // half[64][256]  gelu(x@W1^T+b1)          (C)
#define OFF_RS     (152*1024)  // float[64] row sums                      (C)
#define OFF_RQ     (152*1024 + 256) // float[64] row sumsq                (C)
#define LDS_BYTES  (152*1024 + 512)

// ---- workspace (f16 weights), element offsets -------------------------------
#define WS_WQ  0
#define WS_WK  16384
#define WS_WV  32768
#define WS_W1  49152
#define WS_W2  114688
#define WS_E   180224   // dist_emb padded to [128][64], row 127 = 0

// ---------------------------------------------------------------------------
// Wave32 fragment layouts per CDNA5 ISA 7.12.2:
//  A (16x32 f16): lane L -> row M=L&15, k-half kh=L>>4; k=(i<8?0:16)+kh*8+(i&7)
//  B (32x16 f16): lane L -> col N=L&15, k-half kg=L>>4; k=(i<8?0:16)+kg*8+(i&7)
//  C/D (16x16 f32): lane L -> col N=L&15, reg r -> row M = r + (L>>4)*8
// ---------------------------------------------------------------------------
__device__ __forceinline__ v8f wmma32(v16h a, v16h b, v8f c) {
  return __builtin_amdgcn_wmma_f32_16x16x32_f16(
      false, a, false, b, (short)0, c, false, false);
}

__device__ __forceinline__ v16h ldA(const _Float16* base, int stride,
                                    int m0, int k0, int lane) {
  int m  = lane & 15;
  int kh = lane >> 4;
  const _Float16* p = base + (size_t)(m0 + m) * stride + k0 + kh * 8;
  v8h lo = *(const v8h*)(p);
  v8h hi = *(const v8h*)(p + 16);
  v16h a;
#pragma unroll
  for (int i = 0; i < 8; ++i) { a[i] = lo[i]; a[i + 8] = hi[i]; }
  return a;
}

// B fragment for out = X @ SRC^T : B[k][n] = SRC[n][k], SRC row-major [N x K].
__device__ __forceinline__ v16h ldB(const _Float16* src, int stride,
                                    int n0, int k0, int lane) {
  int n  = lane & 15;
  int kg = lane >> 4;
  const _Float16* p = src + (size_t)(n0 + n) * stride + k0 + kg * 8;
  v8h lo = *(const v8h*)(p);
  v8h hi = *(const v8h*)(p + 16);
  v16h b;
#pragma unroll
  for (int i = 0; i < 8; ++i) { b[i] = lo[i]; b[i + 8] = hi[i]; }
  return b;
}

// ---------------------------------------------------------------------------
__global__ void qf_cvt_f16(const float* __restrict__ src,
                           _Float16* __restrict__ dst, int n, int nsrc) {
  int i = blockIdx.x * blockDim.x + threadIdx.x;
  if (i < n) dst[i] = (i < nsrc) ? (_Float16)src[i] : (_Float16)0.0f;
}

// ---------------------------------------------------------------------------
extern "C" __global__ __launch_bounds__(256, 1)
void qf_layer_kernel(const float* __restrict__ q_repr,
                     const float* __restrict__ im_repr,
                     const float* __restrict__ bq,
                     const float* __restrict__ bk,
                     const float* __restrict__ bv,
                     const float* __restrict__ b1,
                     const float* __restrict__ b2,
                     const float* __restrict__ ln_g,
                     const float* __restrict__ ln_b,
                     const _Float16* __restrict__ wh,
                     float* __restrict__ out) {
  extern __shared__ char smem[];
  _Float16* A16   = (_Float16*)(smem + OFF_A16);
  _Float16* Q16   = (_Float16*)(smem + OFF_Q16);
  _Float16* K16   = (_Float16*)(smem + OFF_K16);
  _Float16* VT16  = (_Float16*)(smem + OFF_VT16);
  float*    S32   = (float*)   (smem + OFF_S32);
  _Float16* P16   = (_Float16*)(smem + OFF_P16);
  _Float16* XCAT  = (_Float16*)(smem + OFF_XCAT);
  _Float16* INTER = (_Float16*)(smem + OFF_INTER);
  float*    RS    = (float*)   (smem + OFF_RS);
  float*    RQ    = (float*)   (smem + OFF_RQ);
  float*    H32   = (float*)   (smem + OFF_A16);   // phase-C alias of [0,64K)

  const int b    = blockIdx.x;
  const int tid  = threadIdx.x;
  const int lane = tid & 31;
  // wave index is truly wave-uniform: make it scalar so tile math lives in SGPRs
  const int wave = __builtin_amdgcn_readfirstlane(tid >> 5);
  const int n_   = lane & 15;
  const int mh_  = lane >> 4;

  // warm L2/WGP$ with FFN weights while phases A/B run
  __builtin_prefetch(wh + WS_W1, 0, 1);
  __builtin_prefetch(wh + WS_W2, 0, 1);

  // ===== Phase A: stage q_repr[b] as f16, compute Q/K/V ====================
  const float* xq = q_repr + (size_t)b * S_ * HID_;
  for (int e = tid; e < S_ * HID_; e += 256) A16[e] = (_Float16)xq[e];
  __syncthreads();

  // 24 column-groups (3 mats x 8 Ntiles) over 8 waves; 4 M-accumulators each.
  for (int i = 0; i < 3; ++i) {
    int g   = wave * 3 + i;                 // scalar
    int mat = g >> 3;
    int nt  = g & 7;
    size_t woff = (mat == 0) ? (size_t)WS_WQ
                : (mat == 1) ? (size_t)WS_WK : (size_t)WS_WV;
    const float* bp = (mat == 0) ? bq : (mat == 1) ? bk : bv;
    v8f a0 = {}, a1 = {}, a2 = {}, a3 = {};
#pragma unroll
    for (int ks = 0; ks < 4; ++ks) {
      v16h bb = ldB(wh + woff, HID_, nt * 16, ks * 32, lane);
      a0 = wmma32(ldA(A16, HID_, 0,  ks * 32, lane), bb, a0);
      a1 = wmma32(ldA(A16, HID_, 16, ks * 32, lane), bb, a1);
      a2 = wmma32(ldA(A16, HID_, 32, ks * 32, lane), bb, a2);
      a3 = wmma32(ldA(A16, HID_, 48, ks * 32, lane), bb, a3);
    }
    int col = nt * 16 + n_;
    float bias = bp[col];
    if (mat == 2) {                          // V stored transposed
#pragma unroll
      for (int r = 0; r < 8; ++r) {
        VT16[col * S_ + ( 0 + mh_ * 8 + r)] = (_Float16)(a0[r] + bias);
        VT16[col * S_ + (16 + mh_ * 8 + r)] = (_Float16)(a1[r] + bias);
        VT16[col * S_ + (32 + mh_ * 8 + r)] = (_Float16)(a2[r] + bias);
        VT16[col * S_ + (48 + mh_ * 8 + r)] = (_Float16)(a3[r] + bias);
      }
    } else {
      _Float16* dst = (mat == 0) ? Q16 : K16;
#pragma unroll
      for (int r = 0; r < 8; ++r) {
        dst[( 0 + mh_ * 8 + r) * HID_ + col] = (_Float16)(a0[r] + bias);
        dst[(16 + mh_ * 8 + r) * HID_ + col] = (_Float16)(a1[r] + bias);
        dst[(32 + mh_ * 8 + r) * HID_ + col] = (_Float16)(a2[r] + bias);
        dst[(48 + mh_ * 8 + r) * HID_ + col] = (_Float16)(a3[r] + bias);
      }
    }
  }
  __syncthreads();

  // ===== Phase B: attention per head ======================================
  for (int h = 0; h < NH_; ++h) {
    // scores = Q_h @ K_h^T   (16 tiles over 8 waves)
    for (int it = 0; it < 2; ++it) {
      int t = wave * 2 + it;
      int mt = t >> 2, nt = t & 3;
      v8f acc = {};
#pragma unroll
      for (int ks = 0; ks < 2; ++ks) {
        v16h a  = ldA(Q16, HID_, mt * 16, h * 64 + ks * 32, lane);
        v16h bb = ldB(K16, HID_, nt * 16, h * 64 + ks * 32, lane);
        acc = wmma32(a, bb, acc);
      }
#pragma unroll
      for (int r = 0; r < 8; ++r)
        S32[(mt * 16 + mh_ * 8 + r) * S_ + nt * 16 + n_] = acc[r];
    }
    __syncthreads();

    // rel-pos bias: T = Q_h @ E^T; scatter S32[l][l+63-j] += T[l][j]
    for (int it = 0; it < 4; ++it) {
      int t = wave * 4 + it;                 // 32 tiles: 4 x 8
      int mt = t >> 3, jt = t & 7;
      v8f acc = {};
#pragma unroll
      for (int ks = 0; ks < 2; ++ks) {
        v16h a  = ldA(Q16, HID_, mt * 16, h * 64 + ks * 32, lane);
        v16h bb = ldB(wh + WS_E, DH_, jt * 16, ks * 32, lane);
        acc = wmma32(a, bb, acc);
      }
      int j = jt * 16 + n_;
#pragma unroll
      for (int r = 0; r < 8; ++r) {
        int M  = mt * 16 + mh_ * 8 + r;
        int rr = M + (MAXP_ - 1) - j;
        if (rr >= 0 && rr < S_) S32[M * S_ + rr] += acc[r];
      }
    }
    __syncthreads();

    // softmax over rows (scale 1/sqrt(64) = 0.125)
    if (tid < S_) {
      float* row = S32 + tid * S_;
      float mx = -3.0e38f;
      for (int r2 = 0; r2 < S_; ++r2) {
        float v = row[r2] * 0.125f;
        row[r2] = v;
        mx = fmaxf(mx, v);
      }
      float sum = 0.0f;
      for (int r2 = 0; r2 < S_; ++r2) {
        float e = __expf(row[r2] - mx);
        row[r2] = e;
        sum += e;
      }
      float inv = 1.0f / sum;
      _Float16* prow = P16 + tid * S_;
      for (int r2 = 0; r2 < S_; ++r2) prow[r2] = (_Float16)(row[r2] * inv);
    }
    __syncthreads();

    // ctx = P @ V_h -> XCAT columns [128 + h*64, 128 + h*64 + 64)
    for (int it = 0; it < 2; ++it) {
      int t = wave * 2 + it;
      int mt = t >> 2, dt = t & 3;
      v8f acc = {};
#pragma unroll
      for (int ks = 0; ks < 2; ++ks) {
        v16h a  = ldA(P16, S_, mt * 16, ks * 32, lane);
        v16h bb = ldB(VT16, S_, h * 64 + dt * 16, ks * 32, lane);
        acc = wmma32(a, bb, acc);
      }
#pragma unroll
      for (int r = 0; r < 8; ++r) {
        int M = mt * 16 + mh_ * 8 + r;
        XCAT[M * FF_ + HID_ + h * 64 + dt * 16 + n_] = (_Float16)acc[r];
      }
    }
    __syncthreads();
  }

  // ===== Phase C: FFN + residual + LayerNorm ===============================
  const float* xi = im_repr + (size_t)b * S_ * HID_;
  for (int e = tid; e < S_ * HID_; e += 256) {
    int m = e >> 7, c = e & 127;
    XCAT[m * FF_ + c] = (_Float16)xi[e];
  }
  if (tid < S_) { RS[tid] = 0.0f; RQ[tid] = 0.0f; }
  __syncthreads();

  // GEMM1: inter = gelu(x @ W1^T + b1); 16 col-groups over 8 waves, 4 M-accs.
  for (int i = 0; i < 2; ++i) {
    int nt = wave * 2 + i;                   // scalar
    v8f a0 = {}, a1 = {}, a2 = {}, a3 = {};
#pragma unroll
    for (int ks = 0; ks < 8; ++ks) {
      v16h bb = ldB(wh + WS_W1, FF_, nt * 16, ks * 32, lane);
      a0 = wmma32(ldA(XCAT, FF_, 0,  ks * 32, lane), bb, a0);
      a1 = wmma32(ldA(XCAT, FF_, 16, ks * 32, lane), bb, a1);
      a2 = wmma32(ldA(XCAT, FF_, 32, ks * 32, lane), bb, a2);
      a3 = wmma32(ldA(XCAT, FF_, 48, ks * 32, lane), bb, a3);
    }
    int col = nt * 16 + n_;
    float bias = b1[col];
#pragma unroll
    for (int r = 0; r < 8; ++r) {
      float u0 = a0[r] + bias, u1 = a1[r] + bias;
      float u2 = a2[r] + bias, u3 = a3[r] + bias;
      INTER[( 0 + mh_ * 8 + r) * FF_ + col] =
          (_Float16)(0.5f * u0 * (1.0f + erff(u0 * 0.70710678f)));
      INTER[(16 + mh_ * 8 + r) * FF_ + col] =
          (_Float16)(0.5f * u1 * (1.0f + erff(u1 * 0.70710678f)));
      INTER[(32 + mh_ * 8 + r) * FF_ + col] =
          (_Float16)(0.5f * u2 * (1.0f + erff(u2 * 0.70710678f)));
      INTER[(48 + mh_ * 8 + r) * FF_ + col] =
          (_Float16)(0.5f * u3 * (1.0f + erff(u3 * 0.70710678f)));
    }
  }
  __syncthreads();

  // GEMM2: h = x + (inter @ W2^T + b2); h kept in LDS (aliases dead QKV bufs)
  for (int i = 0; i < 2; ++i) {
    int nt = wave * 2 + i;
    v8f a0 = {}, a1 = {}, a2 = {}, a3 = {};
#pragma unroll
    for (int ks = 0; ks < 8; ++ks) {
      v16h bb = ldB(wh + WS_W2, FF_, nt * 16, ks * 32, lane);
      a0 = wmma32(ldA(INTER, FF_, 0,  ks * 32, lane), bb, a0);
      a1 = wmma32(ldA(INTER, FF_, 16, ks * 32, lane), bb, a1);
      a2 = wmma32(ldA(INTER, FF_, 32, ks * 32, lane), bb, a2);
      a3 = wmma32(ldA(INTER, FF_, 48, ks * 32, lane), bb, a3);
    }
    int col = nt * 16 + n_;
    float bias = b2[col];
#pragma unroll
    for (int r = 0; r < 8; ++r) {
      int M0 =  0 + mh_ * 8 + r, M1 = 16 + mh_ * 8 + r;
      int M2 = 32 + mh_ * 8 + r, M3 = 48 + mh_ * 8 + r;
      float h0 = a0[r] + bias + (float)XCAT[M0 * FF_ + col];
      float h1 = a1[r] + bias + (float)XCAT[M1 * FF_ + col];
      float h2 = a2[r] + bias + (float)XCAT[M2 * FF_ + col];
      float h3 = a3[r] + bias + (float)XCAT[M3 * FF_ + col];
      H32[M0 * FF_ + col] = h0;
      H32[M1 * FF_ + col] = h1;
      H32[M2 * FF_ + col] = h2;
      H32[M3 * FF_ + col] = h3;
      atomicAdd(&RS[M0], h0); atomicAdd(&RQ[M0], h0 * h0);
      atomicAdd(&RS[M1], h1); atomicAdd(&RQ[M1], h1 * h1);
      atomicAdd(&RS[M2], h2); atomicAdd(&RQ[M2], h2 * h2);
      atomicAdd(&RS[M3], h3); atomicAdd(&RQ[M3], h3 * h3);
    }
  }
  __syncthreads();

  // LayerNorm and write-out
  float* ob = out + (size_t)b * S_ * FF_;
  const float invff = 1.0f / (float)FF_;
  for (int e = tid; e < S_ * FF_; e += 256) {
    int m = e >> 8, c = e & 255;
    float mu   = RS[m] * invff;
    float var  = RQ[m] * invff - mu * mu;
    float rstd = rsqrtf(var + 1e-12f);
    ob[e] = (H32[e] - mu) * rstd * ln_g[c] + ln_b[c];
  }
}

// ---------------------------------------------------------------------------
extern "C" void kernel_launch(void* const* d_in, const int* in_sizes, int n_in,
                              void* d_out, int out_size, void* d_ws, size_t ws_size,
                              hipStream_t stream) {
  (void)in_sizes; (void)n_in; (void)out_size; (void)ws_size;
  const float* im   = (const float*)d_in[0];
  const float* qr   = (const float*)d_in[1];
  const float* Wq   = (const float*)d_in[2];
  const float* bq   = (const float*)d_in[3];
  const float* Wk   = (const float*)d_in[4];
  const float* bk   = (const float*)d_in[5];
  const float* Wv   = (const float*)d_in[6];
  const float* bv   = (const float*)d_in[7];
  const float* dist = (const float*)d_in[8];
  const float* W1   = (const float*)d_in[9];
  const float* b1   = (const float*)d_in[10];
  const float* W2   = (const float*)d_in[11];
  const float* b2   = (const float*)d_in[12];
  const float* lg   = (const float*)d_in[13];
  const float* lb   = (const float*)d_in[14];

  _Float16* wh = (_Float16*)d_ws;

  auto cvt = [&](const float* s, _Float16* dst, int n, int nsrc) {
    qf_cvt_f16<<<(n + 255) / 256, 256, 0, stream>>>(s, dst, n, nsrc);
  };
  cvt(Wq,   wh + WS_WQ, 16384, 16384);
  cvt(Wk,   wh + WS_WK, 16384, 16384);
  cvt(Wv,   wh + WS_WV, 16384, 16384);
  cvt(W1,   wh + WS_W1, 65536, 65536);
  cvt(W2,   wh + WS_W2, 65536, 65536);
  cvt(dist, wh + WS_E,  128 * 64, (2 * MAXP_ - 1) * DH_);  // pad row 127 = 0

  qf_layer_kernel<<<B_, 256, LDS_BYTES, stream>>>(
      qr, im, bq, bk, bv, b1, b2, lg, lb, wh, (float*)d_out);
}
